// sSelfAttentionLayer_68745246540065
// MI455X (gfx1250) — matrix operand order repaired and checked by reading, not compiled, not run on hardware
//
#include <hip/hip_runtime.h>

// ---------------- problem dims ----------------
#define B_    16
#define T_    12
#define NN_   325
#define GP_   64
#define D_    512
#define H_    8
#define HD_   64
#define NTOT_ 389
#define BT_   192
#define FF_   2048
#define NKPM  416   // padded key/token dim (>= 389, multiple of 32)

typedef __attribute__((ext_vector_type(16))) _Float16 v16h;
typedef __attribute__((ext_vector_type(8)))  _Float16 v8h;
typedef __attribute__((ext_vector_type(8)))  float    v8f;
typedef __attribute__((ext_vector_type(4)))  float    v4f;

#define WMMA_F16(Af, Bf, Cf) \
  __builtin_amdgcn_wmma_f32_16x16x32_f16(false, (Af), false, (Bf), (short)0, (Cf), false, false)

// ---------------- CDNA5 async global->LDS copy (ASYNCcnt), with safe fallback ----------------
#if defined(__has_builtin)
#  if __has_builtin(__builtin_amdgcn_global_load_async_to_lds_b128)
#    define ASYNC_COPY 1
#  endif
#endif
#ifndef ASYNC_COPY
#  define ASYNC_COPY 0
#endif

#if ASYNC_COPY
#  if __has_builtin(__builtin_amdgcn_s_wait_asynccnt)
#    define ASYNC_WAIT(n) __builtin_amdgcn_s_wait_asynccnt(n)
#  else
#    define ASYNC_WAIT(n) asm volatile("s_wait_asynccnt " #n ::: "memory")
#  endif
typedef int vi4 __attribute__((__vector_size__(16)));
typedef __attribute__((address_space(1))) vi4 gas_vi4;   // global (AS1)
typedef __attribute__((address_space(3))) vi4 las_vi4;   // LDS (AS3)
__device__ __forceinline__ void async_cp16(const _Float16* g, _Float16* l) {
  __builtin_amdgcn_global_load_async_to_lds_b128((gas_vi4*)g, (las_vi4*)l, 0, 0);
}
#else
#  define ASYNC_WAIT(n) ((void)0)
#endif

// ---------- fragment loaders (16-bit WMMA, wave32; this code's contiguous k order, same for A and B) ----------
// A (16x32, MxK): lane m=lid&15 row; 16 contiguous K at hi*16
__device__ __forceinline__ v16h fragA_ptr(const _Float16* p) {
  v8h lo = *(const v8h*)p;
  v8h hi = *(const v8h*)(p + 8);
  v16h f;
#pragma unroll
  for (int i = 0; i < 8; ++i) { f[i] = lo[i]; f[i + 8] = hi[i]; }
  return f;
}
__device__ __forceinline__ v16h fragA_lds(const _Float16* base, int rstride, int lid) {
  return fragA_ptr(base + (lid & 15) * rstride + (lid >> 4) * 16);
}
// B (32x16, KxN) stored N-major (Bt[n][k]): lane c=lid&15 col; 16 contiguous K at hi*16
__device__ __forceinline__ v16h fragB_ptr(const _Float16* p) {
  v8h lo = *(const v8h*)p;
  v8h hi = *(const v8h*)(p + 8);
  v16h f;
#pragma unroll
  for (int i = 0; i < 8; ++i) { f[i] = lo[i]; f[i + 8] = hi[i]; }
  return f;
}
__device__ __forceinline__ v16h fragB_lds(const _Float16* base, int rstride, int lid) {
  return fragB_ptr(base + (lid & 15) * rstride + (lid >> 4) * 16);
}

// ---------------- weight transpose + f16 convert ----------------
// W: (K x N) f32 row-major  ->  Wt: (N x K) f16 row-major
__global__ __launch_bounds__(256) void transpose_f16(const float* __restrict__ W,
                                                     _Float16* __restrict__ Wt,
                                                     int K, int N) {
  int idx = blockIdx.x * 256 + threadIdx.x;
  if (idx >= N * K) return;
  int n = idx / K, k = idx - n * K;
  Wt[idx] = (_Float16)W[(long)k * N + n];
}

// ---------------- generic WMMA GEMM (templated flags, double-buffered async LDS) ----------------
#define TM 64
#define TN 128
#define TK 32
#define TKP 40  // padded LDS K stride (bank-conflict avoidance)

enum { F_AH = 1, F_RELU = 2, F_OH = 4, F_TRANS = 8 };

template <int FLAGS>
__global__ __launch_bounds__(256) void gemm_wmma(
    const float* __restrict__ A0, const float* __restrict__ A1, int srcThresh,
    long aB0, long aB1,
    const _Float16* __restrict__ Ah, long ahB, int lda,
    const _Float16* __restrict__ Wt,   // (N x K) f16
    const float* __restrict__ bias,    // [N]
    float* __restrict__ Of, _Float16* __restrict__ Oh, long oB, int ldc,
    int M, int K, int Mpad) {
  __shared__ __align__(16) _Float16 sA[2][TM][TKP];
  __shared__ __align__(16) _Float16 sB[2][TN][TKP];

  const int tid = threadIdx.x;
  const int z = blockIdx.z;
  const int m0 = blockIdx.y * TM;
  const int n0 = blockIdx.x * TN;
  const int wv = tid >> 5, lid = tid & 31;
  const int msub = wv >> 1, nsub = wv & 1;

  // stage one K-tile (A + B) into LDS buffer `buf`
  auto stage = [&](int buf, int kt) {
    if constexpr (FLAGS & F_AH) {
      int row = tid >> 2, c8 = tid & 3;
      int ar = m0 + row; if (ar > M - 1) ar = M - 1;
      const _Float16* src = Ah + (long)z * ahB + (long)ar * lda + kt + c8 * 8;
#if ASYNC_COPY
      async_cp16(src, &sA[buf][row][c8 * 8]);
#else
      *(v8h*)&sA[buf][row][c8 * 8] = *(const v8h*)src;
#endif
    } else {
#pragma unroll
      for (int i = 0; i < 2; ++i) {
        int chunk = tid + i * 256;
        int row = chunk >> 3, c4 = chunk & 7;
        int ar = m0 + row; if (ar > M - 1) ar = M - 1;
        const float* base = (ar < srcThresh) ? (A0 + (long)z * aB0) : (A1 + (long)z * aB1);
        v4f d = *(const v4f*)(base + (long)ar * lda + kt + c4 * 4);
#pragma unroll
        for (int q = 0; q < 4; ++q) sA[buf][row][c4 * 4 + q] = (_Float16)d[q];
      }
    }
#pragma unroll
    for (int i = 0; i < 2; ++i) {
      int chunk = tid + i * 256;
      int row = chunk >> 2, c8 = chunk & 3;
      const _Float16* src = Wt + (long)(n0 + row) * K + kt + c8 * 8;
#if ASYNC_COPY
      async_cp16(src, &sB[buf][row][c8 * 8]);
#else
      *(v8h*)&sB[buf][row][c8 * 8] = *(const v8h*)src;
#endif
    }
  };

  v8f acc[4] = {};
  int cur = 0;
  stage(0, 0);

  for (int kt = 0; kt < K; kt += TK) {
    const bool more = (kt + TK) < K;
    if (more) stage(cur ^ 1, kt + TK);   // issue next tile while current computes
#if ASYNC_COPY
    if (more) {
      if constexpr (FLAGS & F_AH) { ASYNC_WAIT(3); } else { ASYNC_WAIT(2); }
    } else {
      ASYNC_WAIT(0);
    }
#endif
    __syncthreads();

    v16h af = fragA_lds(&sA[cur][msub * 16][0], TKP, lid);
#pragma unroll
    for (int j = 0; j < 4; ++j) {
      v16h bf = fragB_lds(&sB[cur][nsub * 64 + j * 16][0], TKP, lid);
      acc[j] = WMMA_F16(af, bf, acc[j]);
    }
    __syncthreads();
    cur ^= 1;
  }

  // ---- epilogue: bias / relu / f16-f32 / transposed ----
  const int c = lid & 15, hb = lid >> 4;
#pragma unroll
  for (int j = 0; j < 4; ++j) {
    int coln = n0 + nsub * 64 + j * 16 + c;
    float bv = bias[coln];
#pragma unroll
    for (int r = 0; r < 8; ++r) {
      int mrow = m0 + msub * 16 + r + 8 * hb;
      float v = acc[j][r] + bv;
      if constexpr (FLAGS & F_RELU) v = fmaxf(v, 0.0f);
      if constexpr (FLAGS & F_TRANS) {
        if (mrow < Mpad)
          Oh[(long)z * oB + (long)coln * ldc + mrow] = (_Float16)((mrow < M) ? v : 0.0f);
      } else {
        if (mrow < M) {
          long idx = (long)z * oB + (long)mrow * ldc + coln;
          if constexpr (FLAGS & F_OH) Oh[idx] = (_Float16)v;
          else                        Of[idx] = v;
        }
      }
    }
  }
}

// ---------------- fused attention (scores -> softmax -> P@V), per (z,h,qtile) ----------------
__global__ __launch_bounds__(128) void attn_wmma(
    const _Float16* __restrict__ Q, long qB, int Nq,
    const _Float16* __restrict__ Kh, long kB, int Nk,
    const _Float16* __restrict__ Vt, long vB, int ldv,   // Vt: (512 x ldv) per slice
    _Float16* __restrict__ O, long oB) {
  extern __shared__ char smem[];
  float*    S = (float*)smem;                          // [4][16][NKPM]
  _Float16* P = (_Float16*)(smem + 4 * 16 * NKPM * 4); // [4][16][NKPM]

  const int z = blockIdx.z, h = blockIdx.y;
  const int wv = threadIdx.x >> 5, lid = threadIdx.x & 31;
  const int q0 = blockIdx.x * 64 + wv * 16;
  const int nkp = (Nk + 31) & ~31;

  float*    Sw = S + wv * 16 * NKPM;
  _Float16* Pw = P + wv * 16 * NKPM;

  const int c  = lid & 15;
  const int hb = lid >> 4;

  // Q fragments for this wave's 16 query rows (K = 0..63 in two chunks)
  int qr = q0 + c; if (qr > Nq - 1) qr = Nq - 1;
  const _Float16* qp = Q + (long)z * qB + (long)qr * D_ + h * HD_ + hb * 16;
  v16h qa0 = fragA_ptr(qp);
  v16h qa1 = fragA_ptr(qp + 32);

  // ---- scores: S = (Q K^T) * 1/sqrt(64) ----
  for (int n0 = 0; n0 < nkp; n0 += 16) {
    int ktok = n0 + c; if (ktok > Nk - 1) ktok = Nk - 1;
    const _Float16* kp = Kh + (long)z * kB + (long)ktok * D_ + h * HD_ + hb * 16;
    v16h kb0 = fragB_ptr(kp);
    v16h kb1 = fragB_ptr(kp + 32);
    v8f s = {};
    s = WMMA_F16(qa0, kb0, s);
    s = WMMA_F16(qa1, kb1, s);
    bool oob = (n0 + c) >= Nk;
#pragma unroll
    for (int r = 0; r < 8; ++r)
      Sw[(r + 8 * hb) * NKPM + n0 + c] = oob ? -1.0e30f : s[r] * 0.125f;
  }
  __syncthreads();

  // ---- softmax: 2 lanes per row (lane L and L+16) ----
  {
    float* srow = &Sw[c * NKPM];
    float mx = -1.0e30f;
    for (int j = hb; j < Nk; j += 2) mx = fmaxf(mx, srow[j]);
    mx = fmaxf(mx, __shfl_xor(mx, 16, 32));
    float sum = 0.0f;
    for (int j = hb; j < Nk; j += 2) { float e = __expf(srow[j] - mx); srow[j] = e; sum += e; }
    sum += __shfl_xor(sum, 16, 32);
    float inv = 1.0f / sum;
    _Float16* prow = &Pw[c * NKPM];
    for (int j = hb; j < nkp; j += 2) prow[j] = (_Float16)((j < Nk) ? srow[j] * inv : 0.0f);
  }
  __syncthreads();

  // ---- O = P @ V  (V pre-transposed: Vt[hd][token], zero-padded tokens) ----
#pragma unroll 1
  for (int nc = 0; nc < 4; ++nc) {
    v8f acc = {};
    for (int kk = 0; kk < nkp; kk += 32) {
      v16h pa = fragA_lds(&Pw[kk], NKPM, lid);
      const _Float16* vp = Vt + (long)z * vB + (long)(h * HD_ + nc * 16 + c) * ldv + kk + hb * 16;
      v16h vb = fragB_ptr(vp);
      acc = WMMA_F16(pa, vb, acc);
    }
#pragma unroll
    for (int r = 0; r < 8; ++r) {
      int row = q0 + r + 8 * hb;
      if (row < Nq)
        O[(long)z * oB + (long)row * D_ + h * HD_ + nc * 16 + c] = (_Float16)acc[r];
    }
  }
}

// ---------------- fused residual + LayerNorm (one wave per 512-wide row) ----------------
// flags bit0: residual b comes from f32 X with token remap; bit1: output f32 to d_out with remap
__global__ __launch_bounds__(256) void ln_fused(
    const _Float16* __restrict__ a,
    const _Float16* __restrict__ bh,
    const float* __restrict__ bf,
    const float* __restrict__ g, const float* __restrict__ beta,
    _Float16* __restrict__ outH, float* __restrict__ outF,
    int rows, int flags) {
  int row = blockIdx.x * 8 + (threadIdx.x >> 5);
  if (row >= rows) return;
  int lid = threadIdx.x & 31;
  int zz = row / NN_, jj = row - zz * NN_;
  long rrow = (long)zz * NTOT_ + GP_ + jj;

  const _Float16* pa = a + (long)row * D_;
  float x[16];
  float s = 0.0f;
#pragma unroll
  for (int i = 0; i < 16; ++i) {
    int ci = lid + i * 32;
    float av = (float)pa[ci];
    float rv = (flags & 1) ? bf[rrow * D_ + ci] : (float)bh[(long)row * D_ + ci];
    x[i] = av + rv;
    s += x[i];
  }
#pragma unroll
  for (int o = 16; o > 0; o >>= 1) s += __shfl_xor(s, o, 32);
  float mean = s * (1.0f / 512.0f);
  float var = 0.0f;
#pragma unroll
  for (int i = 0; i < 16; ++i) { float d = x[i] - mean; var += d * d; }
#pragma unroll
  for (int o = 16; o > 0; o >>= 1) var += __shfl_xor(var, o, 32);
  float rsq = rsqrtf(var * (1.0f / 512.0f) + 1e-5f);
#pragma unroll
  for (int i = 0; i < 16; ++i) {
    int ci = lid + i * 32;
    float y = (x[i] - mean) * rsq * g[ci] + beta[ci];
    if (flags & 2) outF[rrow * D_ + ci] = y;
    else           outH[(long)row * D_ + ci] = (_Float16)y;
  }
}

// ---------------- host orchestration ----------------
template <int FLAGS>
static inline void launch_gemm(const float* A0, const float* A1, int thresh,
                               long aB0, long aB1,
                               const _Float16* Ah, long ahB, int lda,
                               const _Float16* Wt, const float* bias,
                               float* Of, _Float16* Oh, long oB, int ldc,
                               int M, int N, int K, int Mpad, hipStream_t stream) {
  int mspan = (FLAGS & F_TRANS) ? Mpad : M;
  dim3 grid(N / TN, (mspan + TM - 1) / TM, BT_);
  gemm_wmma<FLAGS><<<grid, dim3(256), 0, stream>>>(A0, A1, thresh, aB0, aB1, Ah, ahB, lda,
                                                   Wt, bias, Of, Oh, oB, ldc, M, K, Mpad);
}

extern "C" void kernel_launch(void* const* d_in, const int* in_sizes, int n_in,
                              void* d_out, int out_size, void* d_ws, size_t ws_size,
                              hipStream_t stream) {
  (void)in_sizes; (void)n_in; (void)out_size; (void)ws_size;
  const float* X    = (const float*)d_in[0];
  const float* e_wq = (const float*)d_in[1];  const float* e_bq = (const float*)d_in[2];
  const float* e_wk = (const float*)d_in[3];  const float* e_bk = (const float*)d_in[4];
  const float* e_wv = (const float*)d_in[5];  const float* e_bv = (const float*)d_in[6];
  const float* e_wo = (const float*)d_in[7];  const float* e_bo = (const float*)d_in[8];
  const float* d_wq = (const float*)d_in[9];  const float* d_bq = (const float*)d_in[10];
  const float* d_wk = (const float*)d_in[11]; const float* d_bk = (const float*)d_in[12];
  const float* d_wv = (const float*)d_in[13]; const float* d_bv = (const float*)d_in[14];
  const float* d_wo = (const float*)d_in[15]; const float* d_bo = (const float*)d_in[16];
  const float* ffw1 = (const float*)d_in[17]; const float* ffb1 = (const float*)d_in[18];
  const float* ffw2 = (const float*)d_in[19]; const float* ffb2 = (const float*)d_in[20];
  const float* ln1g = (const float*)d_in[21]; const float* ln1b = (const float*)d_in[22];
  const float* ln2g = (const float*)d_in[23]; const float* ln2b = (const float*)d_in[24];
  float* out = (float*)d_out;

  // workspace carve-up
  char* wp = (char*)d_ws;
  auto carve = [&](size_t bytes) -> char* {
    char* p = wp; wp += (bytes + 255) & ~(size_t)255; return p;
  };
  _Float16* t_ewq = (_Float16*)carve((size_t)D_ * D_ * 2);
  _Float16* t_ewk = (_Float16*)carve((size_t)D_ * D_ * 2);
  _Float16* t_ewv = (_Float16*)carve((size_t)D_ * D_ * 2);
  _Float16* t_ewo = (_Float16*)carve((size_t)D_ * D_ * 2);
  _Float16* t_dwq = (_Float16*)carve((size_t)D_ * D_ * 2);
  _Float16* t_dwk = (_Float16*)carve((size_t)D_ * D_ * 2);
  _Float16* t_dwv = (_Float16*)carve((size_t)D_ * D_ * 2);
  _Float16* t_dwo = (_Float16*)carve((size_t)D_ * D_ * 2);
  _Float16* t_ff1 = (_Float16*)carve((size_t)D_ * FF_ * 2);  // (2048 x 512)
  _Float16* t_ff2 = (_Float16*)carve((size_t)FF_ * D_ * 2);  // (512 x 2048)
  _Float16* encQ  = (_Float16*)carve((size_t)BT_ * GP_ * D_ * 2);
  _Float16* encK  = (_Float16*)carve((size_t)BT_ * NN_ * D_ * 2);
  _Float16* encVt = (_Float16*)carve((size_t)BT_ * D_ * NKPM * 2);
  _Float16* encA  = (_Float16*)carve((size_t)BT_ * GP_ * D_ * 2);
  _Float16* decQ  = (_Float16*)carve((size_t)BT_ * NN_ * D_ * 2);
  _Float16* decK  = (_Float16*)carve((size_t)BT_ * NTOT_ * D_ * 2);
  _Float16* decVt = (_Float16*)carve((size_t)BT_ * D_ * NKPM * 2);
  _Float16* decA  = (_Float16*)carve((size_t)BT_ * NN_ * D_ * 2);
  _Float16* decO  = (_Float16*)carve((size_t)BT_ * NN_ * D_ * 2);
  _Float16* ln1o  = (_Float16*)carve((size_t)BT_ * NN_ * D_ * 2);
  _Float16* hid   = (_Float16*)carve((size_t)BT_ * NN_ * FF_ * 2);
  _Float16* ffo   = (_Float16*)carve((size_t)BT_ * NN_ * D_ * 2);

  // ---- convert + transpose all weights to f16 (N x K) ----
  auto tr = [&](const float* W, _Float16* Wt, int K, int N) {
    int n = N * K;
    transpose_f16<<<(n + 255) / 256, 256, 0, stream>>>(W, Wt, K, N);
  };
  tr(e_wq, t_ewq, D_, D_); tr(e_wk, t_ewk, D_, D_);
  tr(e_wv, t_ewv, D_, D_); tr(e_wo, t_ewo, D_, D_);
  tr(d_wq, t_dwq, D_, D_); tr(d_wk, t_dwk, D_, D_);
  tr(d_wv, t_dwv, D_, D_); tr(d_wo, t_dwo, D_, D_);
  tr(ffw1, t_ff1, D_, FF_); tr(ffw2, t_ff2, FF_, D_);

  const long XB = (long)NTOT_ * D_;
  const int  BIG = 0x7FFFFFFF;

  // ---- encoder MHA: q=glob(64), kv=patch(325) ----
  launch_gemm<F_OH>(X, X, BIG, XB, XB, nullptr, 0, D_, t_ewq, e_bq,
                    nullptr, encQ, (long)GP_ * D_, D_, GP_, D_, D_, 0, stream);
  launch_gemm<F_OH>(X + GP_ * D_, X + GP_ * D_, BIG, XB, XB, nullptr, 0, D_, t_ewk, e_bk,
                    nullptr, encK, (long)NN_ * D_, D_, NN_, D_, D_, 0, stream);
  launch_gemm<F_OH | F_TRANS>(X + GP_ * D_, X + GP_ * D_, BIG, XB, XB, nullptr, 0, D_, t_ewv, e_bv,
                              nullptr, encVt, (long)D_ * NKPM, NKPM, NN_, D_, D_, NKPM, stream);

  const size_t ATT_LDS = (size_t)4 * 16 * NKPM * 4 + (size_t)4 * 16 * NKPM * 2;
  attn_wmma<<<dim3(1, H_, BT_), 128, ATT_LDS, stream>>>(
      encQ, (long)GP_ * D_, GP_, encK, (long)NN_ * D_, NN_,
      encVt, (long)D_ * NKPM, NKPM, encA, (long)GP_ * D_);

  // encoder O-proj writes t directly into d_out tokens [0,64)
  launch_gemm<F_AH>(nullptr, nullptr, 0, 0, 0, encA, (long)GP_ * D_, D_, t_ewo, e_bo,
                    out, nullptr, (long)NTOT_ * D_, D_, GP_, D_, D_, 0, stream);

  // ---- decoder MHA: q=patch(325), kv=x_kv(389) = [t | patch] ----
  launch_gemm<F_OH>(X + GP_ * D_, X + GP_ * D_, BIG, XB, XB, nullptr, 0, D_, t_dwq, d_bq,
                    nullptr, decQ, (long)NN_ * D_, D_, NN_, D_, D_, 0, stream);
  launch_gemm<F_OH>(out, X, GP_, (long)NTOT_ * D_, XB, nullptr, 0, D_, t_dwk, d_bk,
                    nullptr, decK, (long)NTOT_ * D_, D_, NTOT_, D_, D_, 0, stream);
  launch_gemm<F_OH | F_TRANS>(out, X, GP_, (long)NTOT_ * D_, XB, nullptr, 0, D_, t_dwv, d_bv,
                              nullptr, decVt, (long)D_ * NKPM, NKPM, NTOT_, D_, D_, NKPM, stream);

  attn_wmma<<<dim3((NN_ + 63) / 64, H_, BT_), 128, ATT_LDS, stream>>>(
      decQ, (long)NN_ * D_, NN_, decK, (long)NTOT_ * D_, NTOT_,
      decVt, (long)D_ * NKPM, NKPM, decA, (long)NN_ * D_);

  launch_gemm<F_AH | F_OH>(nullptr, nullptr, 0, 0, 0, decA, (long)NN_ * D_, D_, t_dwo, d_bo,
                           nullptr, decO, (long)NN_ * D_, D_, NN_, D_, D_, 0, stream);

  // ---- LN1: ln1 = LN(patch + decO) ----
  int rows = BT_ * NN_;
  ln_fused<<<(rows + 7) / 8, 256, 0, stream>>>(decO, nullptr, X, ln1g, ln1b,
                                               ln1o, nullptr, rows, 1);

  // ---- FF: hid = relu(ln1 @ W1 + b1); ffo = hid @ W2 + b2 ----
  launch_gemm<F_AH | F_OH | F_RELU>(nullptr, nullptr, 0, 0, 0, ln1o, (long)NN_ * D_, D_, t_ff1, ffb1,
                                    nullptr, hid, (long)NN_ * FF_, FF_, NN_, FF_, D_, 0, stream);
  launch_gemm<F_AH | F_OH>(nullptr, nullptr, 0, 0, 0, hid, (long)NN_ * FF_, FF_, t_ff2, ffb2,
                           nullptr, ffo, (long)NN_ * D_, D_, NN_, D_, FF_, 0, stream);

  // ---- LN2: d_out tokens [64,389) = LN(ln1 + ffo) ----
  ln_fused<<<(rows + 7) / 8, 256, 0, stream>>>(ln1o, ffo, nullptr, ln2g, ln2b,
                                               nullptr, out, rows, 2);
}
